// SingleLSTM_2087354106534
// MI455X (gfx1250) — matrix operand order repaired and checked by reading, compiled, not run on hardware
//
#include <hip/hip_runtime.h>

#define TSTEPS 28
#define NINPUT 28
#define HID    128
#define GATES  512   // 4*HID
#define NLAB   10

typedef __attribute__((ext_vector_type(16))) _Float16 v16h;
typedef __attribute__((ext_vector_type(8)))  float    v8f;

// ---------------------------------------------------------------------------
// Weight packing: f32 W_lstm [156][512] -> f16 WMMA *A-operand* blocks of
// W^T in d_ws (D = W^T_tile(16 gate-cols x K) x act(K x 16 batch)).
// Layout (halfs):
//   [0      .. 16384) : Wx^T, 32 ntile blocks of 512 halfs (K padded 28->32)
//   [16384  .. 81920) : Wh^T, 128 (kc,ntile) blocks of 512 halfs
// A-operand layout inside a block: lane l (0..31), half hh (0..15):
//   M(gate col) = ntile*16 + (l&15)
//   k_local     = ((l&16)?16:0)+hh
// ---------------------------------------------------------------------------
__global__ void pack_weights_kernel(const float* __restrict__ W,
                                    _Float16* __restrict__ wp) {
  int idx = blockIdx.x * blockDim.x + threadIdx.x;
  if (idx >= 81920) return;
  int blk  = idx >> 9;      // /512
  int r    = idx & 511;
  int lane = r >> 4;
  int hh   = r & 15;
  int gcol   = (lane & 15);
  int klocal = ((lane & 16) ? 16 : 0) + hh;
  float v;
  if (blk < 32) {                       // Wx^T block
    int ntile = blk;
    int n = ntile * 16 + gcol;
    v = (klocal < NINPUT) ? W[klocal * GATES + n] : 0.0f;
  } else {                              // Wh^T block
    int b2 = blk - 32;
    int kc = b2 >> 5;
    int ntile = b2 & 31;
    int n = ntile * 16 + gcol;
    int k = kc * 32 + klocal;           // 0..127
    v = W[(NINPUT + k) * GATES + n];
  }
  wp[idx] = (_Float16)v;
}

// ---------------------------------------------------------------------------
// Nonlinearities: native CDNA5 TANH transcendental when available.
// ---------------------------------------------------------------------------
#if __has_builtin(__builtin_amdgcn_tanhf)
__device__ __forceinline__ float tanh_f(float x) {
  return __builtin_amdgcn_tanhf(x);
}
#else
__device__ __forceinline__ float tanh_f(float x) {
  return 1.0f - 2.0f * __builtin_amdgcn_rcpf(1.0f + __expf(2.0f * x));
}
#endif
__device__ __forceinline__ float sigmoid_f(float x) {
  return fmaf(0.5f, tanh_f(0.5f * x), 0.5f);
}
__device__ __forceinline__ float sigmoid_b1_f(float x) {   // sigmoid(x + 1.0)
  return fmaf(0.5f, tanh_f(fmaf(0.5f, x, 0.5f)), 0.5f);
}

// ---------------------------------------------------------------------------
// Persistent LSTM, weight-stationary-in-REGISTERS:
//  - 128 threads = 4 waves; block owns 128 batch rows for all 28 steps.
//  - gates^T = W^T x [x;h]^T : A = W^T fragments (40 v16h = 320 VGPRs/wave,
//    loaded once), B = activation fragments (lane = batch column).
//  - Wave w owns gate columns { g*128 + w*32 .. +32 } for g=i,j,f,o, so
//    elementwise c/h updates stay wave-local (c in 128 registers).
//  - h crosses waves via a double-buffered 2x(128x128) f16 LDS tile with a
//    single __syncthreads() per step. Each B fragment feeds 8 WMMA chains.
// ---------------------------------------------------------------------------
__launch_bounds__(128, 1)
__global__ void lstm_kernel(const float* __restrict__ x,
                            const _Float16* __restrict__ wpack,
                            const float* __restrict__ b_lstm,
                            const float* __restrict__ W_fc,
                            const float* __restrict__ b_fc,
                            float* __restrict__ out) {
  extern __shared__ unsigned char smem[];
  _Float16* hb = (_Float16*)smem;     // 2 buffers x 128x128 halfs = 64 KB

  const int tid  = threadIdx.x;
  const int lane = tid & 31;
  const int wave = tid >> 5;
  const int lo   = lane & 15;
  const bool hiH = (lane >= 16);
  const int row0 = blockIdx.x * 128;  // batch rows of this block

  // ---- load this wave's W^T A-fragments into registers (kept resident) ----
  // ntile owned: nt(g,s) = g*8 + wave*2 + s
  const v16h* wp = (const v16h*)wpack;
  v16h wxA[4][2];
  v16h whA[4][2][4];
#pragma unroll
  for (int g = 0; g < 4; ++g)
#pragma unroll
    for (int s = 0; s < 2; ++s) {
      const int nt = g * 8 + wave * 2 + s;
      wxA[g][s] = wp[nt * 32 + lane];
#pragma unroll
      for (int kc = 0; kc < 4; ++kc)
        whA[g][s][kc] = wp[(32 + kc * 32 + nt) * 32 + lane];
    }

  // ---- bias per owned gate column (D row r maps to col base + r (+8)) ----
  float bias_r[4][2][8];
#pragma unroll
  for (int g = 0; g < 4; ++g)
#pragma unroll
    for (int s = 0; s < 2; ++s)
#pragma unroll
      for (int r = 0; r < 8; ++r)
        bias_r[g][s][r] =
            b_lstm[g * HID + wave * 32 + s * 16 + r + (hiH ? 8 : 0)];

  // ---- zero h buffer 0 ----
  {
    float4* hz = (float4*)hb;
    float4 z = make_float4(0.f, 0.f, 0.f, 0.f);
    for (int i = tid; i < 2048; i += 128) hz[i] = z;   // 32 KB
  }
  __syncthreads();

  // c-state: (hidden col = wave*32 + s*16 + r + (hiH?8:0), batch = bt*16+lo)
  float cst[2][8][8];   // [s][bt][r]
#pragma unroll
  for (int s = 0; s < 2; ++s)
#pragma unroll
    for (int b = 0; b < 8; ++b)
#pragma unroll
      for (int r = 0; r < 8; ++r) cst[s][b][r] = 0.0f;

#pragma unroll 1
  for (int t = 0; t < TSTEPS; ++t) {
    const int rb = (t & 1) * 16384;        // read buffer (halfs)
    const int wb = 16384 - rb;             // write buffer

#pragma unroll
    for (int bt = 0; bt < 8; ++bt) {
      // ---- x B-fragment: lane = batch col, halves = K ascending ----
      // lane<16: K 0..15 ; lane>=16: K 16..31 (28..31 zero-padded)
      const float* xr = x + (size_t)(row0 + bt * 16 + lo) * (TSTEPS * NINPUT)
                        + t * NINPUT + (hiH ? 16 : 0);
      float4 f0 = *(const float4*)(xr + 0);
      float4 f1 = *(const float4*)(xr + 4);
      float4 f2 = *(const float4*)(xr + 8);
      float4 f3 = hiH ? make_float4(0.f, 0.f, 0.f, 0.f)
                      : *(const float4*)(xr + 12);
      v16h bxf;
      bxf[0]=(_Float16)f0.x;  bxf[1]=(_Float16)f0.y;  bxf[2]=(_Float16)f0.z;  bxf[3]=(_Float16)f0.w;
      bxf[4]=(_Float16)f1.x;  bxf[5]=(_Float16)f1.y;  bxf[6]=(_Float16)f1.z;  bxf[7]=(_Float16)f1.w;
      bxf[8]=(_Float16)f2.x;  bxf[9]=(_Float16)f2.y;  bxf[10]=(_Float16)f2.z; bxf[11]=(_Float16)f2.w;
      bxf[12]=(_Float16)f3.x; bxf[13]=(_Float16)f3.y; bxf[14]=(_Float16)f3.z; bxf[15]=(_Float16)f3.w;

      // ---- h B-fragments from read buffer (32B contiguous per lane) ----
      v16h bh[4];
#pragma unroll
      for (int kc = 0; kc < 4; ++kc)
        bh[kc] = *(const v16h*)(hb + rb + (bt * 16 + lo) * HID + kc * 32
                                + (hiH ? 16 : 0));

      // ---- 8 accumulator chains: D = W^T x act  (each bh feeds 8 WMMAs) ----
      v8f acc[4][2];
#pragma unroll
      for (int g = 0; g < 4; ++g)
#pragma unroll
        for (int s = 0; s < 2; ++s) {
          v8f a;
#pragma unroll
          for (int r = 0; r < 8; ++r) a[r] = bias_r[g][s][r];
          a = __builtin_amdgcn_wmma_f32_16x16x32_f16(false, wxA[g][s], false, bxf, (short)0, a, false, false);
#pragma unroll
          for (int kc = 0; kc < 4; ++kc)
            a = __builtin_amdgcn_wmma_f32_16x16x32_f16(false, whA[g][s][kc], false, bh[kc], (short)0, a, false, false);
          acc[g][s] = a;
        }

      // ---- elementwise LSTM update; h -> write buffer ----
#pragma unroll
      for (int s = 0; s < 2; ++s) {
#pragma unroll
        for (int r = 0; r < 8; ++r) {
          float gi = sigmoid_f(acc[0][s][r]);
          float gj = tanh_f(acc[1][s][r]);
          float gf = sigmoid_b1_f(acc[2][s][r]);     // sigmoid(f + 1.0)
          float go = sigmoid_f(acc[3][s][r]);
          float c  = fmaf(cst[s][bt][r], gf, gi * gj);
          cst[s][bt][r] = c;
          float h = tanh_f(c) * go;
          const int col = wave * 32 + s * 16 + r + (hiH ? 8 : 0);
          hb[wb + (bt * 16 + lo) * HID + col] = (_Float16)h;
        }
      }
    }
    __syncthreads();   // writes to wb visible before next step reads it
  }

  // ---- final FC: h is in buffer 0 (28 steps => last write was buf 0) ----
  {
    const int rw = tid;                  // 128 threads, 128 rows
    float acc[NLAB];
#pragma unroll
    for (int l = 0; l < NLAB; ++l) acc[l] = b_fc[l];
#pragma unroll 1
    for (int k = 0; k < HID; ++k) {
      float hv = (float)hb[rw * HID + k];
#pragma unroll
      for (int l = 0; l < NLAB; ++l) acc[l] += hv * W_fc[k * NLAB + l];
    }
    float* op = out + (size_t)(row0 + rw) * NLAB;
#pragma unroll
    for (int l = 0; l < NLAB; ++l) op[l] = acc[l];
  }
}

extern "C" void kernel_launch(void* const* d_in, const int* in_sizes, int n_in,
                              void* d_out, int out_size, void* d_ws, size_t ws_size,
                              hipStream_t stream) {
  const float* x      = (const float*)d_in[0];
  const float* W_lstm = (const float*)d_in[1];
  const float* b_lstm = (const float*)d_in[2];
  const float* W_fc   = (const float*)d_in[3];
  const float* b_fc   = (const float*)d_in[4];
  float* out = (float*)d_out;
  _Float16* wpack = (_Float16*)d_ws;   // 163840 bytes used

  const int B = in_sizes[0] / (TSTEPS * NINPUT);   // 32768

  pack_weights_kernel<<<(81920 + 255) / 256, 256, 0, stream>>>(W_lstm, wpack);

  const int blocks = B / 128;                      // 256 blocks, 4 waves each
  const size_t lds = 2 * 128 * HID * sizeof(_Float16);   // 64 KB double buffer
  lstm_kernel<<<blocks, 128, lds, stream>>>(x, wpack, b_lstm, W_fc, b_fc, out);
}
